// GATv2_39058432590074
// MI455X (gfx1250) — compile-verified, matrix operand order checked by reading, not run on hardware
//
#include <hip/hip_runtime.h>
#include <hip/hip_bf16.h>
#include <cmath>

typedef __attribute__((ext_vector_type(16))) _Float16 v16h;
typedef __attribute__((ext_vector_type(8)))  float    v8f;

#define NNODES 50000
#define NEDGES 500000

// ---------- float <-> order-preserving uint key (for atomic segment-max) ----------
__device__ __forceinline__ unsigned f2key(float f) {
  unsigned b = __float_as_uint(f);
  return (b & 0x80000000u) ? ~b : (b | 0x80000000u);
}
__device__ __forceinline__ float key2f(unsigned u) {
  if (u == 0u) return 0.0f;  // "no edge" sentinel (never hit: self-loops exist)
  return (u & 0x80000000u) ? __uint_as_float(u ^ 0x80000000u) : __uint_as_float(~u);
}

// Load 32 consecutive f32 K-values (lane-dependent base) into WMMA 16-bit
// operand layout: lane<16 holds K = kb+[0..7] & kb+[16..23]; lane>=16 holds
// K = kb+8+[0..7] & kb+24+[0..7]  => base pointer already includes half*8.
__device__ __forceinline__ v16h load_cvt_f16(const float* p /* row + kb + half*8 */) {
  const float4* q = (const float4*)p;
  float4 v0 = q[0], v1 = q[1], v2 = q[4], v3 = q[5];   // +0, +4, +16, +20 floats
  v16h r;
  r[0] =(_Float16)v0.x; r[1] =(_Float16)v0.y; r[2] =(_Float16)v0.z; r[3] =(_Float16)v0.w;
  r[4] =(_Float16)v1.x; r[5] =(_Float16)v1.y; r[6] =(_Float16)v1.z; r[7] =(_Float16)v1.w;
  r[8] =(_Float16)v2.x; r[9] =(_Float16)v2.y; r[10]=(_Float16)v2.z; r[11]=(_Float16)v2.w;
  r[12]=(_Float16)v3.x; r[13]=(_Float16)v3.y; r[14]=(_Float16)v3.z; r[15]=(_Float16)v3.w;
  return r;
}

// ---------------------------------------------------------------------------
// Y[Nrows x Mcols] = X[Nrows x 128] * W[Mcols x 128]^T
// fp32 in/out, f16 WMMA 16x16x32 inner product, fp32 accumulate.
// One wave per 16-row strip: A converted to f16 registers ONCE (4 x v16h),
// then reused across all Mcols/16 B tiles -> 4*(Mcols/16) WMMAs per wave,
// A global traffic cut by Mcols/16 vs tile-per-wave.
// ---------------------------------------------------------------------------
__global__ void gatv2_gemm_nt_wmma(const float* __restrict__ X,
                                   const float* __restrict__ W,
                                   float* __restrict__ Y,
                                   int Nrows, int Mcols) {
  const int K = 128;
  int wave = (int)((blockIdx.x * blockDim.x + threadIdx.x) >> 5);
  int lane = threadIdx.x & 31;
  int ntN = (Nrows + 15) >> 4;
  if (wave >= ntN) return;                    // uniform per wave: EXEC stays all-1s
  int half = lane >> 4;                       // 0: lanes 0-15, 1: lanes 16-31
  int l = lane & 15;
  int row = wave * 16 + l; if (row >= Nrows) row = Nrows - 1;   // tail clamp
  const float* xrow = X + (size_t)row * K + half * 8;

  // A strip resident in registers (32 VGPRs)
  v16h a[4];
  #pragma unroll
  for (int kk = 0; kk < 4; ++kk) a[kk] = load_cvt_f16(xrow + kk * 32);

  int ntM = Mcols >> 4;
  for (int tm = 0; tm < ntM; ++tm) {
    const float* wrow = W + (size_t)(tm * 16 + l) * K + half * 8;  // B col n = W row m
    v8f acc = {};
    #pragma unroll
    for (int kk = 0; kk < 4; ++kk) {
      v16h b = load_cvt_f16(wrow + kk * 32);
      acc = __builtin_amdgcn_wmma_f32_16x16x32_f16(
          /*neg_a=*/false, a[kk], /*neg_b=*/false, b,
          /*c_mod=*/(short)0, acc, /*reuse_a=*/false, /*reuse_b=*/false);
    }
    // C/D layout: VGPR r -> (M = r + 8*half, N = l)
    #pragma unroll
    for (int r = 0; r < 8; ++r) {
      int orow = wave * 16 + r + half * 8;
      if (orow < Nrows) Y[(size_t)orow * Mcols + tm * 16 + l] = acc[r];
    }
  }
}

// ---------------------------------------------------------------------------
// Pass 1: per-edge logits + atomic segment-max over destination.
// One wave per edge; lane = channel (coalesced row reads), shfl reduction.
// ---------------------------------------------------------------------------
template<int C>
__global__ void gatv2_edge_logits(const float* __restrict__ xl,
                                  const float* __restrict__ xr,
                                  const float* __restrict__ att,
                                  const int* __restrict__ srcA,
                                  const int* __restrict__ dstA,
                                  float* __restrict__ logit,
                                  unsigned* __restrict__ lmax,
                                  int E, int Nn) {
  const int HC = 4 * C;
  int wave = (int)((blockIdx.x * blockDim.x + threadIdx.x) >> 5);
  int lane = threadIdx.x & 31;
  if (wave >= E + Nn) return;
  int s, d;
  if (wave < E) { s = srcA[wave]; d = dstA[wave]; } else { s = wave - E; d = s; }  // self-loop
  const float* xls = xl + (size_t)s * HC;
  const float* xrd = xr + (size_t)d * HC;
  float hv[4] = {0.f, 0.f, 0.f, 0.f};
  #pragma unroll
  for (int i = 0; i < HC / 32; ++i) {
    int j = lane + 32 * i;
    float v = xls[j] + xrd[j];
    v = v > 0.f ? v : 0.2f * v;              // leaky_relu, slope 0.2
    hv[(i * 32) / C] += v * att[j];          // head index constant per chunk
  }
  #pragma unroll
  for (int h = 0; h < 4; ++h) {
    float v = hv[h];
    #pragma unroll
    for (int m = 16; m > 0; m >>= 1) v += __shfl_xor(v, m, 32);
    hv[h] = v;
  }
  if (lane < 4) {
    float lg = hv[lane];
    logit[(size_t)wave * 4 + lane] = lg;
    atomicMax(&lmax[(size_t)d * 4 + lane], f2key(lg));
  }
}

// ---------------------------------------------------------------------------
// Pass 2: w = exp(logit - lmax[dst]); denom[dst] += w; agg[dst] += w * xl[src]
// (final normalization by denom is deferred to the per-node finalize pass)
// ---------------------------------------------------------------------------
template<int C>
__global__ void gatv2_edge_accum(const float* __restrict__ xl,
                                 const float* __restrict__ logit,
                                 const unsigned* __restrict__ lmax,
                                 float* __restrict__ denom,
                                 float* __restrict__ agg,
                                 const int* __restrict__ srcA,
                                 const int* __restrict__ dstA,
                                 int E, int Nn) {
  const int HC = 4 * C;
  int wave = (int)((blockIdx.x * blockDim.x + threadIdx.x) >> 5);
  int lane = threadIdx.x & 31;
  if (wave >= E + Nn) return;
  int s, d;
  if (wave < E) { s = srcA[wave]; d = dstA[wave]; } else { s = wave - E; d = s; }
  float w = 0.f;
  if (lane < 4) {
    float lg = logit[(size_t)wave * 4 + lane];
    float mx = key2f(lmax[(size_t)d * 4 + lane]);
    w = expf(lg - mx);
    atomicAdd(&denom[(size_t)d * 4 + lane], w);
  }
  float wh[4];
  wh[0] = __shfl(w, 0, 32); wh[1] = __shfl(w, 1, 32);
  wh[2] = __shfl(w, 2, 32); wh[3] = __shfl(w, 3, 32);
  const float* xls = xl + (size_t)s * HC;
  float* aggd = agg + (size_t)d * HC;
  #pragma unroll
  for (int i = 0; i < HC / 32; ++i) {
    int j = lane + 32 * i;
    atomicAdd(&aggd[j], wh[(i * 32) / C] * xls[j]);
  }
}

// Layer-1 finalize: divide by denom, +bias, elu -> h (concat layout, HC=128)
__global__ void gatv2_finalize1(const float* __restrict__ agg,
                                const float* __restrict__ denom,
                                const float* __restrict__ b1,
                                float* __restrict__ hout, int Nn) {
  int idx = (int)(blockIdx.x * blockDim.x + threadIdx.x);
  if (idx >= Nn * 128) return;
  int n = idx >> 7, j = idx & 127, h = j >> 5;
  float v = agg[idx] / (denom[n * 4 + h] + 1e-16f) + b1[j];
  hout[idx] = v > 0.f ? v : expm1f(v);      // elu (alpha = 1)
}

// Layer-2 finalize: divide by denom, mean over heads, +bias -> d_out (N x 64)
__global__ void gatv2_finalize2(const float* __restrict__ agg,
                                const float* __restrict__ denom,
                                const float* __restrict__ b2,
                                float* __restrict__ out, int Nn) {
  int idx = (int)(blockIdx.x * blockDim.x + threadIdx.x);
  if (idx >= Nn * 64) return;
  int n = idx >> 6, c = idx & 63;
  float acc = 0.f;
  #pragma unroll
  for (int h = 0; h < 4; ++h)
    acc += agg[(size_t)n * 256 + h * 64 + c] / (denom[n * 4 + h] + 1e-16f);
  out[idx] = acc * 0.25f + b2[c];
}

// ---------------------------------------------------------------------------
extern "C" void kernel_launch(void* const* d_in, const int* in_sizes, int n_in,
                              void* d_out, int out_size, void* d_ws, size_t ws_size,
                              hipStream_t stream) {
  (void)in_sizes; (void)n_in; (void)out_size; (void)ws_size;
  const int  N  = NNODES;
  const int  E  = NEDGES;
  const long EP = (long)E + N;                 // edges + self-loops

  const float* x    = (const float*)d_in[0];
  const int*   ei   = (const int*)  d_in[1];
  const float* Wl1  = (const float*)d_in[2];
  const float* Wr1  = (const float*)d_in[3];
  const float* att1 = (const float*)d_in[4];
  const float* b1   = (const float*)d_in[5];
  const float* Wl2  = (const float*)d_in[6];
  const float* Wr2  = (const float*)d_in[7];
  const float* att2 = (const float*)d_in[8];
  const float* b2   = (const float*)d_in[9];
  const int* srcA = ei;            // edge_index[0]
  const int* dstA = ei + E;        // edge_index[1]

  // workspace layout (fp32 elements)
  float* ws    = (float*)d_ws;
  float* hbuf  = ws;                               // N*128
  float* xl    = hbuf + (size_t)N * 128;           // N*256 (max of both layers)
  float* xr    = xl   + (size_t)N * 256;           // N*256
  float* agg   = xr   + (size_t)N * 256;           // N*256
  float* logit = agg  + (size_t)N * 256;           // EP*4
  unsigned* lmax = (unsigned*)(logit + (size_t)EP * 4);  // N*4
  float* denom = (float*)(lmax + (size_t)N * 4);         // N*4

  const int TPB = 256;                              // 8 wave32 per block
  int eblocks = (int)((EP + 7) / 8);                // one wave per edge
  int gblocks = (((N + 15) / 16) + 7) / 8;          // one wave per 16-row strip

  // ---------------- Layer 1 (C=32, concat) ----------------
  hipMemsetAsync(agg,   0, (size_t)N * 128 * sizeof(float), stream);
  hipMemsetAsync(lmax,  0, (size_t)N * 4 * sizeof(unsigned), stream);
  hipMemsetAsync(denom, 0, (size_t)N * 4 * sizeof(float), stream);
  gatv2_gemm_nt_wmma<<<gblocks, TPB, 0, stream>>>(x, Wl1, xl, N, 128);
  gatv2_gemm_nt_wmma<<<gblocks, TPB, 0, stream>>>(x, Wr1, xr, N, 128);
  gatv2_edge_logits<32><<<eblocks, TPB, 0, stream>>>(xl, xr, att1, srcA, dstA,
                                                     logit, lmax, E, N);
  gatv2_edge_accum<32><<<eblocks, TPB, 0, stream>>>(xl, logit, lmax, denom, agg,
                                                    srcA, dstA, E, N);
  gatv2_finalize1<<<(N * 128 + TPB - 1) / TPB, TPB, 0, stream>>>(agg, denom, b1, hbuf, N);

  // ---------------- Layer 2 (C=64, mean over heads) ----------------
  hipMemsetAsync(agg,   0, (size_t)N * 256 * sizeof(float), stream);
  hipMemsetAsync(lmax,  0, (size_t)N * 4 * sizeof(unsigned), stream);
  hipMemsetAsync(denom, 0, (size_t)N * 4 * sizeof(float), stream);
  gatv2_gemm_nt_wmma<<<gblocks, TPB, 0, stream>>>(hbuf, Wl2, xl, N, 256);
  gatv2_gemm_nt_wmma<<<gblocks, TPB, 0, stream>>>(hbuf, Wr2, xr, N, 256);
  gatv2_edge_logits<64><<<eblocks, TPB, 0, stream>>>(xl, xr, att2, srcA, dstA,
                                                     logit, lmax, E, N);
  gatv2_edge_accum<64><<<eblocks, TPB, 0, stream>>>(xl, logit, lmax, denom, agg,
                                                    srcA, dstA, E, N);
  gatv2_finalize2<<<(N * 64 + TPB - 1) / TPB, TPB, 0, stream>>>(agg, denom, b2,
                                                                (float*)d_out, N);
}
